// _PointnetSAModuleBase_54133767799200
// MI455X (gfx1250) — matrix operand order repaired and checked by reading, not compiled, not run on hardware
//
#include <hip/hip_runtime.h>

typedef _Float16 half_t;
typedef __attribute__((ext_vector_type(16))) _Float16 v16h;
typedef __attribute__((ext_vector_type(8)))  _Float16 v8h;
typedef __attribute__((ext_vector_type(8)))  float    v8f;
typedef __attribute__((ext_vector_type(4)))  unsigned int u32x4;
typedef __attribute__((ext_vector_type(8)))  int          i32x8;
typedef __attribute__((ext_vector_type(4)))  int          i32x4;

#define N_PTS   16384
#define B_SZ    4
#define NPOINT_ 1024
#define NSAMPLE_ 32
#define R2      0.04f
#define CIN     64
#define KPAD    96

// ---------------------------------------------------------------------------
// Kernel 1: furthest point sampling. One block per batch, 1024 threads,
// 16 points per thread kept in VGPRs; argmax via wave shuffles + LDS tree.
// ---------------------------------------------------------------------------
__global__ __launch_bounds__(1024) void fps_kernel(const float* __restrict__ xyz,
                                                   int* __restrict__ cidx)
{
  const int b = blockIdx.x;
  const int t = threadIdx.x;
  const int lane = t & 31, wave = t >> 5;
  const float* xb = xyz + (size_t)b * N_PTS * 3;

  float px[16], py[16], pz[16], dst[16];
#pragma unroll
  for (int j = 0; j < 16; ++j) {
    int i = t + j * 1024;
    px[j] = xb[i*3+0]; py[j] = xb[i*3+1]; pz[j] = xb[i*3+2];
    dst[j] = 1e10f;
  }

  __shared__ float rbest[32];
  __shared__ int   ridx[32];
  __shared__ int   sfar;

  int far = 0;
  for (int it = 0; it < NPOINT_; ++it) {
    if (t == 0) cidx[b*NPOINT_ + it] = far;
    float cx = xb[far*3+0], cy = xb[far*3+1], cz = xb[far*3+2];
    float best = -1.f; int bi = 0;
#pragma unroll
    for (int j = 0; j < 16; ++j) {
      float dx = px[j]-cx, dy = py[j]-cy, dz = pz[j]-cz;
      float d = dx*dx + dy*dy + dz*dz;
      dst[j] = fminf(dst[j], d);
      if (dst[j] > best) { best = dst[j]; bi = t + j*1024; }
    }
#pragma unroll
    for (int off = 16; off > 0; off >>= 1) {
      float ob = __shfl_down(best, off);
      int   oi = __shfl_down(bi,   off);
      if (ob > best || (ob == best && oi < bi)) { best = ob; bi = oi; }
    }
    if (lane == 0) { rbest[wave] = best; ridx[wave] = bi; }
    __syncthreads();
    if (t < 32) {
      best = rbest[t]; bi = ridx[t];
#pragma unroll
      for (int off = 16; off > 0; off >>= 1) {
        float ob = __shfl_down(best, off);
        int   oi = __shfl_down(bi,   off);
        if (ob > best || (ob == best && oi < bi)) { best = ob; bi = oi; }
      }
      if (t == 0) sfar = bi;
    }
    __syncthreads();
    far = sfar;
    __syncthreads();
  }
}

// ---------------------------------------------------------------------------
// Kernel 2: ball query (first-32-by-index semantics) + gather + fp16 pack.
// One wave32 per center; writes a K-padded (96) fp16 row per sample point.
// ---------------------------------------------------------------------------
__global__ __launch_bounds__(256) void group_kernel(const float* __restrict__ xyz,
                                                    const float* __restrict__ feats,
                                                    const int* __restrict__ cidx,
                                                    half_t* __restrict__ Xg,
                                                    float* __restrict__ new_xyz)
{
  const int t = threadIdx.x, lane = t & 31, wave = t >> 5;
  const int gid = blockIdx.x * 8 + wave;          // 0..4095
  const int b = gid >> 10;
  const float* xb = xyz + (size_t)b * N_PTS * 3;

  __shared__ int sidx[8][NSAMPLE_];

  const int ci = cidx[gid];
  const float cx = xb[ci*3+0], cy = xb[ci*3+1], cz = xb[ci*3+2];
  if (lane < 3) new_xyz[gid*3 + lane] = (lane == 0) ? cx : (lane == 1 ? cy : cz);

  int cnt = 0;
  for (int base = 0; base < N_PTS && cnt < NSAMPLE_; base += 32) {
    int i = base + lane;
    float dx = xb[i*3+0]-cx, dy = xb[i*3+1]-cy, dz = xb[i*3+2]-cz;
    bool inb = (dx*dx + dy*dy + dz*dz) < R2;
    unsigned m = (unsigned)__ballot(inb);
    int rank = __popc(m & ((1u << lane) - 1u));
    if (inb && (cnt + rank) < NSAMPLE_) sidx[wave][cnt + rank] = i;
    cnt += __popc(m);
    if (cnt > NSAMPLE_) cnt = NSAMPLE_;
  }

  int myidx;
  if (cnt == 0)            myidx = N_PTS - 1;
  else if (lane < cnt)     myidx = sidx[wave][lane];
  else                     myidx = sidx[wave][0];

  alignas(16) half_t buf[KPAD];
  buf[0] = (half_t)(xb[myidx*3+0] - cx);
  buf[1] = (half_t)(xb[myidx*3+1] - cy);
  buf[2] = (half_t)(xb[myidx*3+2] - cz);
  const float* fb = feats + (size_t)b * CIN * N_PTS + myidx;
#pragma unroll
  for (int c = 0; c < CIN; ++c) buf[3+c] = (half_t)fb[(size_t)c * N_PTS];
#pragma unroll
  for (int k = 3 + CIN; k < KPAD; ++k) buf[k] = (half_t)0.f;

  half_t* dstp = Xg + (size_t)(gid * NSAMPLE_ + lane) * KPAD;
#pragma unroll
  for (int j = 0; j < KPAD/8; ++j) ((v8h*)dstp)[j] = ((const v8h*)buf)[j];
}

// ---------------------------------------------------------------------------
// WMMA fragment helpers, per CDNA5 ISA 16-bit A/B VGPR layouts.
// ---------------------------------------------------------------------------
__device__ __forceinline__ v16h load_A_frag(const half_t* W, int ldk, int mtile,
                                            int kc, int lane) {
  const half_t* p = W + (mtile*16 + (lane & 15)) * ldk + kc*32 + ((lane & 16) ? 8 : 0);
  v8h lo = *(const v8h*)p;
  v8h hi = *(const v8h*)(p + 16);
  v16h f;
#pragma unroll
  for (int j = 0; j < 8; ++j) { f[j] = lo[j]; f[8+j] = hi[j]; }
  return f;
}

__device__ __forceinline__ v16h load_B_frag(const half_t* X, int ldk, int kc, int lane) {
  const half_t* p = X + (lane & 15) * ldk + kc*32 + ((lane & 16) ? 8 : 0);
  v8h lo = *(const v8h*)p;
  v8h hi = *(const v8h*)(p + 16);
  v16h f;
#pragma unroll
  for (int j = 0; j < 8; ++j) { f[j] = lo[j]; f[8+j] = hi[j]; }
  return f;
}

__device__ __forceinline__ v8f wmma_f16(v16h a, v16h b, v8f c) {
  return __builtin_amdgcn_wmma_f32_16x16x32_f16(false, a, false, b, (short)0, c,
                                                false, false);
}

// ---------------------------------------------------------------------------
// Kernel 3: fused 3-layer pointwise MLP (WMMA f16->f32) + 32-sample max-pool.
// The 256x96 fp16 input tile is DMA'd into LDS by the Tensor Data Mover
// (tensor_load_to_lds, TENSORcnt) while the block converts weights.
// ---------------------------------------------------------------------------
__global__ __launch_bounds__(256) void mlp_kernel(
    const half_t* __restrict__ Xg,
    const float* __restrict__ W1, const float* __restrict__ s1, const float* __restrict__ b1,
    const float* __restrict__ W2, const float* __restrict__ s2, const float* __restrict__ b2,
    const float* __restrict__ W3, const float* __restrict__ s3, const float* __restrict__ b3,
    float* __restrict__ out_feat)
{
  extern __shared__ half_t smem[];
  half_t* XgL = smem;                 // 256 x 96 input tile (TDM destination)
  half_t* W1p = XgL + 256*96;         // 64 x 96
  half_t* W2p = W1p + 64*96;          // 64 x 64
  half_t* W3p = W2p + 64*64;          // 128 x 64
  half_t* X1  = W3p + 128*64;         // 256 x 64, point-major
  half_t* X2  = X1  + 256*64;         // 256 x 64, point-major

  const int t = threadIdx.x, lane = t & 31, wave = t >> 5;

  // ---- TDM: DMA the block's 256x96 fp16 tile from global into LDS ----
  if (wave == 0) {
    const unsigned lds_base =
        (unsigned)__builtin_amdgcn_groupstaticsize();      // XgL at dynamic-LDS base
    const unsigned long long ga =
        (unsigned long long)(const void*)(Xg + (size_t)blockIdx.x * 256 * KPAD);

    u32x4 g0;
    g0[0] = 1u;                                            // count=1 (valid D#)
    g0[1] = lds_base;                                      // lds_addr [63:32]
    g0[2] = (unsigned)(ga & 0xffffffffu);                  // global_addr [95:64]
    g0[3] = (unsigned)((ga >> 32) & 0x01ffffffu)           // global_addr [120:96]
          | 0x80000000u;                                   // type=2 at [127:126]

    i32x8 g1;
    g1[0] = 0x00010000;          // workgroup_mask=0, data_size=1 (2 bytes)
    g1[1] = (int)(96u << 16);    // tensor_dim0[15:0] at [63:48]
    g1[2] = (int)(256u << 16);   // tensor_dim0 hi=0, tensor_dim1[15:0] at [95:80]
    g1[3] = (int)(96u << 16);    // tensor_dim1 hi=0, tile_dim0=96 at [127:112]
    g1[4] = 256;                 // tile_dim1=256 at [143:128], tile_dim2=0
    g1[5] = 96;                  // tensor_dim0_stride = 96 elements
    g1[6] = 0;                   // stride hi, tensor_dim1_stride lo = 0
    g1[7] = 0;

    i32x4 g2 = {0, 0, 0, 0};     // unused higher dims (tile_dim3=0)
    i32x4 g3 = {0, 0, 0, 0};
    i32x8 g4 = {0, 0, 0, 0, 0, 0, 0, 0};

    __builtin_amdgcn_tensor_load_to_lds(g0, g1, g2, g3, g4, 0);
  }

  // Weight conversion overlaps with the TDM transfer.
  for (int i = t; i < 64*96; i += 256) {
    int r = i / 96, c = i % 96;
    W1p[i] = (c < 67) ? (half_t)W1[r*67 + c] : (half_t)0.f;
  }
  for (int i = t; i < 64*64; i += 256)  W2p[i] = (half_t)W2[i];
  for (int i = t; i < 128*64; i += 256) W3p[i] = (half_t)W3[i];

  if (wave == 0) __builtin_amdgcn_s_wait_tensorcnt(0);
  __syncthreads();

  // ---- layer 1: (64x96) @ (96x256), K padded with zeros beyond 67 ----
  for (int tt = wave; tt < 4*16; tt += 8) {
    int mt = tt >> 4, nt = tt & 15;
    v8f acc = {};
#pragma unroll
    for (int kc = 0; kc < 3; ++kc) {
      v16h a  = load_A_frag(W1p, KPAD, mt, kc, lane);
      v16h bf = load_B_frag(XgL + nt*16*KPAD, KPAD, kc, lane);
      acc = wmma_f16(a, bf, acc);
    }
    int n = lane & 15, mb = (lane & 16) ? 8 : 0, ch0 = mt*16 + mb;
    v8h h;
#pragma unroll
    for (int r = 0; r < 8; ++r) {
      float v = acc[r]*s1[ch0+r] + b1[ch0+r];
      h[r] = (half_t)(v > 0.f ? v : 0.f);
    }
    *(v8h*)(X1 + (nt*16 + n)*64 + ch0) = h;
  }
  __syncthreads();

  // ---- layer 2: (64x64) @ (64x256) ----
  for (int tt = wave; tt < 4*16; tt += 8) {
    int mt = tt >> 4, nt = tt & 15;
    v8f acc = {};
#pragma unroll
    for (int kc = 0; kc < 2; ++kc) {
      v16h a  = load_A_frag(W2p, 64, mt, kc, lane);
      v16h bf = load_B_frag(X1 + nt*16*64, 64, kc, lane);
      acc = wmma_f16(a, bf, acc);
    }
    int n = lane & 15, mb = (lane & 16) ? 8 : 0, ch0 = mt*16 + mb;
    v8h h;
#pragma unroll
    for (int r = 0; r < 8; ++r) {
      float v = acc[r]*s2[ch0+r] + b2[ch0+r];
      h[r] = (half_t)(v > 0.f ? v : 0.f);
    }
    *(v8h*)(X2 + (nt*16 + n)*64 + ch0) = h;
  }
  __syncthreads();

  // ---- layer 3: (128x64) @ (64x256) fused with 32-sample max-pool ----
  const int gbase = blockIdx.x * 8;
  for (int tt = wave; tt < 8*8; tt += 8) {
    int mt = tt >> 3, g = tt & 7;
    v8f acc0 = {}, acc1 = {};
#pragma unroll
    for (int kc = 0; kc < 2; ++kc) {
      v16h a   = load_A_frag(W3p, 64, mt, kc, lane);
      v16h bf0 = load_B_frag(X2 + (2*g  )*16*64, 64, kc, lane);
      v16h bf1 = load_B_frag(X2 + (2*g+1)*16*64, 64, kc, lane);
      acc0 = wmma_f16(a, bf0, acc0);
      acc1 = wmma_f16(a, bf1, acc1);
    }
    int mb = (lane & 16) ? 8 : 0, ch0 = mt*16 + mb;
    float vals[8];
#pragma unroll
    for (int r = 0; r < 8; ++r) {
      float v0 = acc0[r]*s3[ch0+r] + b3[ch0+r]; v0 = v0 > 0.f ? v0 : 0.f;
      float v1 = acc1[r]*s3[ch0+r] + b3[ch0+r]; v1 = v1 > 0.f ? v1 : 0.f;
      vals[r] = fmaxf(v0, v1);
    }
#pragma unroll
    for (int off = 1; off < 16; off <<= 1) {
#pragma unroll
      for (int r = 0; r < 8; ++r)
        vals[r] = fmaxf(vals[r], __shfl_xor(vals[r], off));
    }
    if ((lane & 15) == 0) {
      int sg = gbase + g, bb = sg >> 10, s = sg & 1023;
      float* o = out_feat + ((size_t)bb*128 + ch0)*1024 + s;
#pragma unroll
      for (int r = 0; r < 8; ++r) o[(size_t)r*1024] = vals[r];
    }
  }
}

// ---------------------------------------------------------------------------
extern "C" void kernel_launch(void* const* d_in, const int* in_sizes, int n_in,
                              void* d_out, int out_size, void* d_ws, size_t ws_size,
                              hipStream_t stream) {
  const float* xyz   = (const float*)d_in[0];
  const float* feats = (const float*)d_in[1];
  const float* W1 = (const float*)d_in[2];
  const float* s1 = (const float*)d_in[3];
  const float* b1 = (const float*)d_in[4];
  const float* W2 = (const float*)d_in[5];
  const float* s2 = (const float*)d_in[6];
  const float* b2 = (const float*)d_in[7];
  const float* W3 = (const float*)d_in[8];
  const float* s3 = (const float*)d_in[9];
  const float* b3 = (const float*)d_in[10];

  float* out      = (float*)d_out;
  float* new_xyz  = out;                       // (4,1024,3)
  float* new_feat = out + B_SZ*NPOINT_*3;      // (4,128,1024)

  int*    cidx = (int*)d_ws;                               // 4*1024 ints
  half_t* Xg   = (half_t*)((char*)d_ws + 32768);           // 131072 x 96 f16

  fps_kernel<<<B_SZ, 1024, 0, stream>>>(xyz, cidx);
  group_kernel<<<512, 256, 0, stream>>>(xyz, feats, cidx, Xg, new_xyz);

  const size_t shmem =
      (size_t)(256*96 + 64*96 + 64*64 + 128*64 + 2*256*64) * sizeof(half_t); // 148 KB
  (void)hipFuncSetAttribute(reinterpret_cast<const void*>(mlp_kernel),
                            hipFuncAttributeMaxDynamicSharedMemorySize, (int)shmem);
  mlp_kernel<<<512, 256, shmem, stream>>>(Xg, W1, s1, b1, W2, s2, b2, W3, s3, b3,
                                          new_feat);
}
